// MpCosineLayer_41077067219609
// MI455X (gfx1250) — matrix operand order repaired and checked by reading, not compiled, not run on hardware
//
#include <hip/hip_runtime.h>
#include <cmath>

typedef _Float16 v8h  __attribute__((ext_vector_type(8)));
typedef _Float16 v16h __attribute__((ext_vector_type(16)));
typedef float    v8f  __attribute__((ext_vector_type(8)));
typedef float    v4f  __attribute__((ext_vector_type(4)));

constexpr int Bn = 64, T1 = 128, T2 = 160, Dd = 768, Mm = 64;

constexpr int KW  = 32;            // K per v_wmma_f32_16x16x32_f16
constexpr int KCH = 64;            // staged K-chunk = 2 WMMA k-steps per barrier
constexpr int NCH = Dd / KCH;      // 12 chunks
constexpr int LDB = Dd + 8;        // Bmat LDS stride (halves), conflict-free b128
constexpr int LDA = KCH + 8;       // A-chunk LDS stride (halves) = 72

constexpr unsigned BS_HALVES = Mm * LDB;                         // 49664
constexpr unsigned AS_HALVES = T2 * LDA;                         // 11520
constexpr unsigned AS0_OFF   = BS_HALVES * 2;                    // byte offsets in LDS
constexpr unsigned AS1_OFF   = (BS_HALVES + AS_HALVES) * 2;
constexpr unsigned LDS_BYTES = (BS_HALVES + 2 * AS_HALVES) * 2;  // 145408 B

// ---------- fp32 -> f16 bulk convert; NT loads (read-once), RT stores (L2-resident) ----
__global__ void cvt_f32_f16(const float* __restrict__ src,
                            _Float16* __restrict__ dst, int n8) {
  int i = blockIdx.x * blockDim.x + threadIdx.x;
  if (i < n8) {
    v4f a = __builtin_nontemporal_load((const v4f*)src + 2 * i);
    v4f b = __builtin_nontemporal_load((const v4f*)src + 2 * i + 1);
    v8h r = {(_Float16)a.x, (_Float16)a.y, (_Float16)a.z, (_Float16)a.w,
             (_Float16)b.x, (_Float16)b.y, (_Float16)b.z, (_Float16)b.w};
    ((v8h*)dst)[i] = r;
  }
}

// gfx1250 async copy global->LDS, tracked by ASYNCcnt (no VGPR round-trip).
__device__ __forceinline__ void async_b128(uint32_t lds_byte_off, const void* gaddr) {
  asm volatile("global_load_async_to_lds_b128 %0, %1, off"
               :: "v"(lds_byte_off), "v"(gaddr) : "memory");
}
__device__ __forceinline__ void wait_async0() {
  asm volatile("s_wait_asynccnt 0" ::: "memory");
}

// Stage v2[b][0:160][kc*64 : +64] into LDS asynchronously. 1280 x 16B, 5/thread.
__device__ __forceinline__ void stageA_async(uint32_t dstOff,
                                             const _Float16* __restrict__ v2b,
                                             int kc, int tid) {
#pragma unroll
  for (int n = 0; n < 5; ++n) {
    int o = tid + n * 256;          // uniform bound: no exec masking
    int r = o >> 3;                 // row 0..159
    int c = (o & 7) * 8;            // half-col 0..56
    async_b128(dstOff + (uint32_t)(r * LDA + c) * 2,
               v2b + (size_t)r * Dd + kc * KCH + c);
  }
}

__device__ __forceinline__ float fast_tanh(float x) {
#if __has_builtin(__builtin_amdgcn_tanhf)
  return __builtin_amdgcn_tanhf(x);   // gfx1250 v_tanh_f32
#else
  return tanhf(x);
#endif
}

// One workgroup per (b,i): Out[j,m] = v2[b](160x768) x Bmat(768x64),
// Bmat[d][m] = v1[b,i,d]*kernel[m,d] built once in LDS.
__global__ __launch_bounds__(256)
void mpcos_wmma(const float* __restrict__ v1,
                const _Float16* __restrict__ v2h,
                const _Float16* __restrict__ kh,
                float* __restrict__ out) {
  extern __shared__ _Float16 smem[];
  _Float16* Bs  = smem;                     // [Mm][LDB]
  _Float16* As0 = smem + BS_HALVES;         // [T2][LDA] double-buffered
  _Float16* As1 = As0 + AS_HALVES;

  const int tid  = threadIdx.x;
  const int lane = tid & 31;                // wave32
  const int wid  = tid >> 5;                // 0..7
  const int mt   = wid & 3;                 // m-tile (4x16 = 64)
  const int jg   = wid >> 2;                // j-group: tiles jg, jg+2, ..., jg+8
  const int m0   = mt * 16;
  const int l15  = lane & 15;
  const int hi   = lane >> 4;

  const int bi = blockIdx.x;
  const int b  = bi / T1;
  const int i  = bi % T1;

  const float*    v1row = v1  + ((size_t)b * T1 + i) * Dd;
  const _Float16* v2b   = v2h + (size_t)b * T2 * Dd;

  // kick off async staging of A chunk 0, overlapped with Bmat build
  stageA_async(AS0_OFF, v2b, 0, tid);

  // ---- build Bmat, stored m-major (B operand is column-major): Bs[m*LDB + d] ----
  for (int idx = tid; idx < Mm * (Dd / 8); idx += 256) {
    int m  = idx / (Dd / 8);
    int d0 = (idx % (Dd / 8)) * 8;
    v8h kv = *(const v8h*)(kh + (size_t)m * Dd + d0);
    float4 a0 = *(const float4*)(v1row + d0);
    float4 a1 = *(const float4*)(v1row + d0 + 4);
    float af[8] = {a0.x, a0.y, a0.z, a0.w, a1.x, a1.y, a1.z, a1.w};
    v8h r;
#pragma unroll
    for (int q = 0; q < 8; ++q) r[q] = (_Float16)(af[q] * (float)kv[q]);
    *(v8h*)(Bs + (size_t)m * LDB + d0) = r;
  }

  wait_async0();
  __syncthreads();

  v8f acc[5];
  {
    v8f z = {0.f, 0.f, 0.f, 0.f, 0.f, 0.f, 0.f, 0.f};
#pragma unroll
    for (int t = 0; t < 5; ++t) acc[t] = z;
  }

  const int kbA = hi * 8;    // A frag: lanes>=16 hold K = 8..15 / 24..31
  const int kbB = hi * 8;    // B frag: lanes>=16 hold K = 8..15 / 24..31

  for (int kc = 0; kc < NCH; ++kc) {
    const _Float16* Acur = (kc & 1) ? As1 : As0;
    if (kc + 1 < NCH)
      stageA_async((kc & 1) ? AS0_OFF : AS1_OFF, v2b, kc + 1, tid);

#pragma unroll
    for (int ks = 0; ks < 2; ++ks) {
      const int k0 = kc * KCH + ks * KW;
      union F { v16h v; v8h h[2]; };

      // load ALL fragments first, then back-to-back WMMAs
      F bf;
      const _Float16* bp = Bs + (size_t)(m0 + l15) * LDB + k0 + kbB;
      bf.h[0] = *(const v8h*)(bp);
      bf.h[1] = *(const v8h*)(bp + 16);

      F af[5];
#pragma unroll
      for (int t = 0; t < 5; ++t) {
        const int j0 = (jg + 2 * t) * 16;
        const _Float16* ap = Acur + (j0 + l15) * LDA + ks * KW + kbA;
        af[t].h[0] = *(const v8h*)(ap);
        af[t].h[1] = *(const v8h*)(ap + 16);
      }
#pragma unroll
      for (int t = 0; t < 5; ++t)
        acc[t] = __builtin_amdgcn_wmma_f32_16x16x32_f16(
            false, af[t].v, false, bf.v, (short)0, acc[t], false, false);
    }

    wait_async0();     // staged chunk kc+1 landed (hidden behind 10 WMMAs)
    __syncthreads();
  }

  // ---- epilogue: v_tanh_f32 + coalesced NON-TEMPORAL stores (keep L2 for inputs) ----
  const size_t outbase = (size_t)bi * T2 * Mm;
#pragma unroll
  for (int t = 0; t < 5; ++t) {
    const int j0 = (jg + 2 * t) * 16;
#pragma unroll
    for (int r = 0; r < 8; ++r) {
      float y = fast_tanh(acc[t][r]);
      __builtin_nontemporal_store(
          y, out + outbase + (size_t)(j0 + r + hi * 8) * Mm + m0 + l15);
    }
  }
}

extern "C" void kernel_launch(void* const* d_in, const int* in_sizes, int n_in,
                              void* d_out, int out_size, void* d_ws, size_t ws_size,
                              hipStream_t stream) {
  const float* v1 = (const float*)d_in[0];
  const float* v2 = (const float*)d_in[1];
  const float* kn = (const float*)d_in[2];
  float* out = (float*)d_out;

  _Float16* v2h = (_Float16*)d_ws;                    // 64*160*768 halves
  _Float16* kh  = v2h + (size_t)Bn * T2 * Dd;         // 64*768 halves

  const int n8v2 = Bn * T2 * Dd / 8;
  const int n8k  = Mm * Dd / 8;
  cvt_f32_f16<<<(n8v2 + 255) / 256, 256, 0, stream>>>(v2, v2h, n8v2);
  cvt_f32_f16<<<(n8k + 255) / 256, 256, 0, stream>>>(kn, kh, n8k);

  mpcos_wmma<<<dim3(Bn * T1), dim3(256), LDS_BYTES, stream>>>(v1, v2h, kh, out);
}
